// GATConv_pr_14998025798501
// MI455X (gfx1250) — compile-verified, matrix operand order checked by reading, not run on hardware
//
#include <hip/hip_runtime.h>

// ---------------------------------------------------------------------------
// GAT pipeline for MI455X (gfx1250, wave32).
//   N=100000 nodes, E=800000 edges, G=64 graphs, D_IN=H=128, D_DOC=256, D_OUT=64
// L2-resident working set; CSR-by-dst build; per-node wave32 online-softmax
// aggregation (no feature atomics); bf16 WMMA (f32 accumulate) node GEMMs with
// pre-swizzled LDS fragments; async global->LDS tile staging (ASYNCcnt).
// ---------------------------------------------------------------------------

typedef __attribute__((ext_vector_type(16))) __bf16 v16bf;
typedef __attribute__((ext_vector_type(8)))  float  v8f;
typedef int v4i_vs __attribute__((vector_size(16)));            // int4, gcc-style
typedef __attribute__((address_space(1))) v4i_vs* gptr_v4i;     // global int4*
typedef __attribute__((address_space(3))) v4i_vs* lptr_v4i;     // LDS int4*

#define HD 128          // hidden / feature dim (fixed by the model)
#define EPS 1e-5f
#define SLOPE 0.2f

#if defined(__has_builtin)
#if __has_builtin(__builtin_amdgcn_global_load_async_to_lds_b128)
#define HAVE_ASYNC_LDS 1
#endif
#endif
#ifndef HAVE_ASYNC_LDS
#define HAVE_ASYNC_LDS 0
#endif

// ---------------------------------------------------------------------------
// 0. init: zero degree array + stats block (re-run every call: graph replay)
// ---------------------------------------------------------------------------
__global__ __launch_bounds__(256) void init_kernel(int* __restrict__ deg,
                                                   float* __restrict__ stats,
                                                   int n, int nstats) {
  int i = blockIdx.x * 256 + threadIdx.x;
  if (i < n)      deg[i]   = 0;
  if (i < nstats) stats[i] = 0.0f;
}

// ---------------------------------------------------------------------------
// 1. CSR build: degree histogram -> exclusive scan -> fill
// ---------------------------------------------------------------------------
__global__ __launch_bounds__(256) void deg_kernel(const int* __restrict__ ei,
                                                  int* __restrict__ deg, int E_) {
  int e = blockIdx.x * 256 + threadIdx.x;
  if (e < E_) atomicAdd(&deg[ei[E_ + e]], 1);   // ei[1][e] = dst
}

__global__ __launch_bounds__(1024) void scan_kernel(const int* __restrict__ deg,
                                                    int* __restrict__ rowp,
                                                    int* __restrict__ cursor, int n) {
  __shared__ int buf[1024];
  __shared__ int carry;
  const int tid = threadIdx.x;
  if (tid == 0) carry = 0;
  __syncthreads();
  for (int base = 0; base < n; base += 1024) {
    int i = base + tid;
    int v = (i < n) ? deg[i] : 0;
    buf[tid] = v;
    __syncthreads();
    #pragma unroll
    for (int off = 1; off < 1024; off <<= 1) {   // Hillis-Steele inclusive scan
      int t = (tid >= off) ? buf[tid - off] : 0;
      __syncthreads();
      buf[tid] += t;
      __syncthreads();
    }
    int c    = carry;
    int excl = buf[tid] - v;
    if (i < n) { rowp[i] = c + excl; cursor[i] = c + excl; }
    int total = buf[1023];
    __syncthreads();
    if (tid == 0) carry = c + total;
    __syncthreads();
  }
  if (tid == 0) rowp[n] = carry;
}

__global__ __launch_bounds__(256) void fill_kernel(const int* __restrict__ ei,
                                                   int* __restrict__ cursor,
                                                   int* __restrict__ colx, int E_) {
  int e = blockIdx.x * 256 + threadIdx.x;
  if (e >= E_) return;
  int src = ei[e];
  int dst = ei[E_ + e];
  int pos = atomicAdd(&cursor[dst], 1);
  colx[pos] = src;
}

// ---------------------------------------------------------------------------
// 2. WMMA GEMM:  Hout[N x 128] = X[N x 128] @ W[128 x 128]
//    256 threads = 8 waves; block tile 64 rows x 128 cols.
//    Wave (rg,ch): rows [16*rg,16*rg+16), cols [64*ch, 64*ch+64) -> 4 C-frags.
//    W staged PRE-SWIZZLED: fragment (kt,ct,lane) contiguous 32 B in LDS
//    (2x ds_load_b128, no repack). X tile staged f32 via
//    GLOBAL_LOAD_ASYNC_TO_LDS_B128 (ASYNCcnt), converted to bf16 at assembly.
// ---------------------------------------------------------------------------
__global__ __launch_bounds__(256) void gemm_node_kernel(const float* __restrict__ X,
                                                        const float* __restrict__ Wf,
                                                        float* __restrict__ Hout,
                                                        int nRows) {
  __shared__ __bf16 sWswz[32 * 32 * 16];   // 32 frags x 32 lanes x 16 elems = 32 KB
  __shared__ float  sXf[64 * HD];          // 64-row f32 tile = 32 KB
  const int tid     = threadIdx.x;
  const int rowBase = blockIdx.x * 64;

  // ---- stage W pre-swizzled (ISA 7.12.2 16-bit B layout baked in) ----
  for (int d = tid; d < 32 * 32 * 16; d += 256) {
    int i    = d & 15;
    int lane = (d >> 4) & 31;
    int frag = d >> 9;                      // kt*8 + ct
    int ct   = frag & 7;
    int kt   = frag >> 3;
    int half = lane >> 4;
    int koff = (i & 7) + ((i & 8) ? 16 : 0) + half * 8;
    sWswz[d] = (__bf16)Wf[(kt * 32 + koff) * HD + ct * 16 + (lane & 15)];
  }

  // ---- stage X tile (f32); OOB rows clamped (their outputs are masked) ----
#if HAVE_ASYNC_LDS
  #pragma unroll
  for (int j = 0; j < 8; ++j) {
    int idx = j * 256 + tid;                // 16-byte chunk id, 2048 total
    int r   = idx >> 5;                     // 32 chunks per 512 B row
    int c4  = (idx & 31) * 4;
    int gr  = rowBase + r;
    if (gr >= nRows) gr = nRows - 1;
    __builtin_amdgcn_global_load_async_to_lds_b128(
        (gptr_v4i)(X + (size_t)gr * HD + c4),
        (lptr_v4i)&sXf[r * HD + c4],
        0, 0);
  }
  asm volatile("s_wait_asynccnt 0x0" ::: "memory");
#else
  for (int d = tid; d < 64 * HD; d += 256) {
    int r = d >> 7, gr = rowBase + r;
    if (gr >= nRows) gr = nRows - 1;
    sXf[d] = X[(size_t)gr * HD + (d & 127)];
  }
#endif
  __syncthreads();

  const int lane = tid & 31;
  const int wave = tid >> 5;
  const int rg   = wave >> 1;              // row group (16 rows)
  const int ch   = wave & 1;               // column half (64 cols)
  const int half = lane >> 4;
  const int l16  = lane & 15;
  const int aRow = rg * 16 + l16;

  v8f c[4] = {};

  #pragma unroll
  for (int kt = 0; kt < 4; ++kt) {
    const int K0 = kt * 32;
    // A frag: two contiguous 32 B chunks per lane (ISA 16-bit A layout), cvt f32->bf16
    v8f lo = *(const v8f*)&sXf[aRow * HD + K0 + half * 8];
    v8f hi = *(const v8f*)&sXf[aRow * HD + K0 + 16 + half * 8];
    v16bf a;
    #pragma unroll
    for (int i = 0; i < 8; ++i) { a[i] = (__bf16)lo[i]; a[8 + i] = (__bf16)hi[i]; }
    #pragma unroll
    for (int ct = 0; ct < 4; ++ct) {
      const v16bf b =
          *(const v16bf*)&sWswz[((kt * 8 + ch * 4 + ct) * 32 + lane) * 16];
      c[ct] = __builtin_amdgcn_wmma_f32_16x16x32_bf16(
          false, a, false, b, (short)0, c[ct], false, false);
    }
  }

  // C layout: VGPR r -> M = r + 8*half, N = 16*ct + l16 (within wave tile)
  if (rowBase + 64 <= nRows) {             // fast path: no per-element guards
    #pragma unroll
    for (int ct = 0; ct < 4; ++ct) {
      #pragma unroll
      for (int r = 0; r < 8; ++r) {
        Hout[(size_t)(rowBase + rg * 16 + r + half * 8) * HD +
             ch * 64 + ct * 16 + l16] = c[ct][r];
      }
    }
  } else {
    #pragma unroll
    for (int ct = 0; ct < 4; ++ct) {
      #pragma unroll
      for (int r = 0; r < 8; ++r) {
        int row = rowBase + rg * 16 + r + half * 8;
        if (row < nRows)
          Hout[(size_t)row * HD + ch * 64 + ct * 16 + l16] = c[ct][r];
      }
    }
  }
}

// ---------------------------------------------------------------------------
// 3. Attention scores: es[i] = h[i].a_src, ed[i] = h[i].a_dst (wave per node)
// ---------------------------------------------------------------------------
__global__ __launch_bounds__(256) void att_score_kernel(const float* __restrict__ Hm,
                                                        const float* __restrict__ as_,
                                                        const float* __restrict__ ad_,
                                                        float* __restrict__ es,
                                                        float* __restrict__ ed, int n) {
  const int lane = threadIdx.x & 31;
  const int node = blockIdx.x * 8 + (threadIdx.x >> 5);
  if (node >= n) return;
  const float* hrow = Hm + (size_t)node * HD;
  float ps = 0.0f, pd = 0.0f;
  #pragma unroll
  for (int k = 0; k < 4; ++k) {
    float v = hrow[lane + 32 * k];
    ps += v * as_[lane + 32 * k];
    pd += v * ad_[lane + 32 * k];
  }
  #pragma unroll
  for (int off = 16; off > 0; off >>= 1) {
    ps += __shfl_xor(ps, off, 32);
    pd += __shfl_xor(pd, off, 32);
  }
  if (lane == 0) { es[node] = ps; ed[node] = pd; }
}

// ---------------------------------------------------------------------------
// 4. GAT aggregation: wave32 per dst node, online softmax over CSR neighbors
//    plus implicit self-loop. No feature atomics. Epilogue: +bias, relu.
// ---------------------------------------------------------------------------
__global__ __launch_bounds__(256) void gat_agg_kernel(const float* __restrict__ Hm,
                                                      const float* __restrict__ es,
                                                      const float* __restrict__ ed,
                                                      const int* __restrict__ rowp,
                                                      const int* __restrict__ colx,
                                                      const float* __restrict__ bias,
                                                      float* __restrict__ Out, int n) {
  const int lane = threadIdx.x & 31;
  const int node = blockIdx.x * 8 + (threadIdx.x >> 5);
  if (node >= n) return;

  const float edi = ed[node];
  float e0 = es[node] + edi;                 // self loop
  e0 = (e0 > 0.0f) ? e0 : SLOPE * e0;
  float m = e0, s = 1.0f;

  float acc[4];
  const float* hrow = Hm + (size_t)node * HD;
  #pragma unroll
  for (int k = 0; k < 4; ++k) acc[k] = hrow[lane + 32 * k];

  const int beg = rowp[node], end = rowp[node + 1];
  for (int j = beg; j < end; ++j) {
    int src = colx[j];
    if (j + 1 < end)                         // CDNA5 global_prefetch_b8 one ahead
      __builtin_prefetch(Hm + (size_t)colx[j + 1] * HD + lane, 0, 3);
    float e = es[src] + edi;
    e = (e > 0.0f) ? e : SLOPE * e;
    const float* hs = Hm + (size_t)src * HD;
    if (e > m) {                             // wave-uniform branch
      float scl = __expf(m - e);
      s = s * scl + 1.0f;
      #pragma unroll
      for (int k = 0; k < 4; ++k) acc[k] = acc[k] * scl + hs[lane + 32 * k];
      m = e;
    } else {
      float w = __expf(e - m);
      s += w;
      #pragma unroll
      for (int k = 0; k < 4; ++k) acc[k] += w * hs[lane + 32 * k];
    }
  }
  const float inv = 1.0f / s;
  float* orow = Out + (size_t)node * HD;
  #pragma unroll
  for (int k = 0; k < 4; ++k) {
    float v = acc[k] * inv + bias[lane + 32 * k];
    orow[lane + 32 * k] = fmaxf(v, 0.0f);
  }
}

// ---------------------------------------------------------------------------
// 5. BatchNorm stats over N (channel = threadIdx.x, contiguous node chunks)
// ---------------------------------------------------------------------------
__global__ __launch_bounds__(128) void bn_stats_kernel(const float* __restrict__ Hin,
                                                       float* __restrict__ gsum,
                                                       float* __restrict__ gsq, int n) {
  const int c    = threadIdx.x;
  const int rows = (n + gridDim.x - 1) / gridDim.x;
  const int r0   = blockIdx.x * rows;
  const int r1   = (r0 + rows < n) ? (r0 + rows) : n;
  float sm = 0.0f, sq = 0.0f;
  for (int r = r0; r < r1; ++r) {
    float v = Hin[(size_t)r * HD + c];
    sm += v; sq += v * v;
  }
  atomicAdd(&gsum[c], sm);
  atomicAdd(&gsq[c], sq);
}

// ---------------------------------------------------------------------------
// 6. BN-apply + relu + mean-pool accumulate (batch sorted -> register
//    accumulation per graph, flush on graph change; ~few k atomics total)
// ---------------------------------------------------------------------------
__global__ __launch_bounds__(128) void bn_pool_kernel(const float* __restrict__ Hin,
                                                      const int* __restrict__ batch,
                                                      const float* __restrict__ gsum,
                                                      const float* __restrict__ gsq,
                                                      const float* __restrict__ gam,
                                                      const float* __restrict__ bet,
                                                      float* __restrict__ psum,
                                                      float* __restrict__ cnt, int n) {
  const int c    = threadIdx.x;
  const float fn = (float)n;
  const float mu  = gsum[c] / fn;
  const float var = gsq[c] / fn - mu * mu;          // biased, matches reference
  const float sc  = rsqrtf(var + EPS) * gam[c];
  const float sh  = bet[c] - mu * sc;

  const int rows = (n + gridDim.x - 1) / gridDim.x;
  const int r0   = blockIdx.x * rows;
  const int r1   = (r0 + rows < n) ? (r0 + rows) : n;

  float acc = 0.0f, cc = 0.0f;
  int curg = -1;
  for (int r = r0; r < r1; ++r) {
    int g = batch[r];                                // uniform across block
    if (g != curg) {
      if (curg >= 0) {
        atomicAdd(&psum[curg * HD + c], acc);
        if (c == 0) atomicAdd(&cnt[curg], cc);
      }
      acc = 0.0f; cc = 0.0f; curg = g;
    }
    float v = Hin[(size_t)r * HD + c] * sc + sh;
    acc += fmaxf(v, 0.0f);
    cc  += 1.0f;
  }
  if (curg >= 0) {
    atomicAdd(&psum[curg * HD + c], acc);
    if (c == 0) atomicAdd(&cnt[curg], cc);
  }
}

// ---------------------------------------------------------------------------
// 7. Head stage 1: pooled | relu(doc GEMM) -> concat -> BN over 64 rows
// ---------------------------------------------------------------------------
__global__ __launch_bounds__(256) void head1_kernel(const float* __restrict__ psum,
                                                    const float* __restrict__ cnt,
                                                    const float* __restrict__ doc,
                                                    const float* __restrict__ docW,
                                                    const float* __restrict__ docb,
                                                    const float* __restrict__ bg,
                                                    const float* __restrict__ bb,
                                                    float* __restrict__ fbn) {
  __shared__ float f[64 * 256];                      // 64 KB
  const int tid = threadIdx.x;
  for (int idx = tid; idx < 64 * 128; idx += 256) {
    int g = idx >> 7, c = idx & 127;
    f[g * 256 + c] = psum[idx] / fmaxf(cnt[g], 1.0f);
  }
  for (int idx = tid; idx < 64 * 128; idx += 256) {
    int g = idx >> 7, c = idx & 127;
    float d = docb[c];
    for (int k = 0; k < 256; ++k) d += doc[g * 256 + k] * docW[k * 128 + c];
    f[g * 256 + 128 + c] = fmaxf(d, 0.0f);
  }
  __syncthreads();
  // BN over 64 rows, 256 channels (one channel per thread)
  float sm = 0.0f, sq = 0.0f;
  for (int g = 0; g < 64; ++g) { float v = f[g * 256 + tid]; sm += v; sq += v * v; }
  float mu  = sm / 64.0f;
  float var = sq / 64.0f - mu * mu;
  float sc  = rsqrtf(var + EPS) * bg[tid];
  float sh  = bb[tid] - mu * sc;
  for (int g = 0; g < 64; ++g) fbn[g * 256 + tid] = f[g * 256 + tid] * sc + sh;
}

// 8. Head stage 2: f2 = relu(fbn @ fus_W + fus_b)   [64 x 128]
__global__ __launch_bounds__(256) void head2_kernel(const float* __restrict__ fbn,
                                                    const float* __restrict__ fusW,
                                                    const float* __restrict__ fusb,
                                                    float* __restrict__ f2) {
  int idx = blockIdx.x * 256 + threadIdx.x;
  if (idx >= 64 * 128) return;
  int g = idx >> 7, c = idx & 127;
  float d = fusb[c];
  for (int k = 0; k < 256; ++k) d += fbn[g * 256 + k] * fusW[k * 128 + c];
  f2[idx] = fmaxf(d, 0.0f);
}

// 9. Head stage 3: task (64x64) then time (64x1), concatenated into d_out
__global__ __launch_bounds__(256) void head3_kernel(const float* __restrict__ f2,
                                                    const float* __restrict__ taskW,
                                                    const float* __restrict__ taskb,
                                                    const float* __restrict__ timeW,
                                                    const float* __restrict__ timeb,
                                                    float* __restrict__ out) {
  int idx = blockIdx.x * 256 + threadIdx.x;
  if (idx < 64 * 64) {
    int g = idx >> 6, o = idx & 63;
    float d = taskb[o];
    for (int k = 0; k < 128; ++k) d += f2[g * 128 + k] * taskW[k * 64 + o];
    out[idx] = d;
  } else if (idx < 64 * 64 + 64) {
    int g = idx - 64 * 64;
    float d = timeb[0];
    for (int k = 0; k < 128; ++k) d += f2[g * 128 + k] * timeW[k];
    out[64 * 64 + g] = d;
  }
}

// ---------------------------------------------------------------------------
// Host: workspace layout + launch sequence (all on `stream`, capture-safe)
// ---------------------------------------------------------------------------
extern "C" void kernel_launch(void* const* d_in, const int* in_sizes, int n_in,
                              void* d_out, int out_size, void* d_ws, size_t ws_size,
                              hipStream_t stream) {
  const float* x     = (const float*)d_in[0];
  const int*   ei    = (const int*)  d_in[1];
  const int*   batch = (const int*)  d_in[2];
  const float* doc   = (const float*)d_in[3];
  const float* W1    = (const float*)d_in[4];
  const float* as1   = (const float*)d_in[5];
  const float* ad1   = (const float*)d_in[6];
  const float* b1    = (const float*)d_in[7];
  const float* W2    = (const float*)d_in[8];
  const float* as2   = (const float*)d_in[9];
  const float* ad2   = (const float*)d_in[10];
  const float* b2    = (const float*)d_in[11];
  const float* bn2g  = (const float*)d_in[12];
  const float* bn2b  = (const float*)d_in[13];
  const float* docW  = (const float*)d_in[14];
  const float* docb  = (const float*)d_in[15];
  const float* bnfg  = (const float*)d_in[16];
  const float* bnfb  = (const float*)d_in[17];
  const float* fusW  = (const float*)d_in[18];
  const float* fusb  = (const float*)d_in[19];
  const float* taskW = (const float*)d_in[20];
  const float* taskb = (const float*)d_in[21];
  const float* timeW = (const float*)d_in[22];
  const float* timeb = (const float*)d_in[23];

  const int N = in_sizes[2];          // batch has N entries
  const int E = in_sizes[1] / 2;      // edge_index is [2, E]

  // ---- workspace layout ----
  float* bufA  = (float*)d_ws;                         // N*128 (h = X@W)
  float* bufB  = bufA + (size_t)N * HD;                // N*128 (aggregated)
  float* es    = bufB + (size_t)N * HD;                // N
  float* ed    = es + N;                               // N
  float* stats = ed + N;                               // 128+128+8192+64 = 8512
  float* gsum  = stats;
  float* gsq   = stats + 128;
  float* psum  = stats + 256;                          // 64*128
  float* cnt   = psum + 64 * HD;                       // 64
  float* fbn   = stats + 8512;                         // 64*256
  float* f2    = fbn + 64 * 256;                       // 64*128
  int*   deg   = (int*)(f2 + 64 * 128);                // N
  int*   curs  = deg + N;                              // N
  int*   rowp  = curs + N;                             // N+1
  int*   colx  = rowp + (N + 1);                       // E

  const int NSTATS = 8512;
  dim3 blk256(256), blk128(128), blk1024(1024);

  // 0-1: scratch init + CSR build (reused by both layers)
  init_kernel<<<dim3((N + 255) / 256), blk256, 0, stream>>>(deg, stats, N, NSTATS);
  deg_kernel <<<dim3((E + 255) / 256), blk256, 0, stream>>>(ei, deg, E);
  scan_kernel<<<dim3(1), blk1024, 0, stream>>>(deg, rowp, curs, N);
  fill_kernel<<<dim3((E + 255) / 256), blk256, 0, stream>>>(ei, curs, colx, E);

  const dim3 gGemm((N + 63) / 64);
  const dim3 gWave((N + 7) / 8);

  // layer 1
  gemm_node_kernel<<<gGemm, blk256, 0, stream>>>(x, W1, bufA, N);
  att_score_kernel<<<gWave, blk256, 0, stream>>>(bufA, as1, ad1, es, ed, N);
  gat_agg_kernel  <<<gWave, blk256, 0, stream>>>(bufA, es, ed, rowp, colx, b1, bufB, N);

  // layer 2
  gemm_node_kernel<<<gGemm, blk256, 0, stream>>>(bufB, W2, bufA, N);
  att_score_kernel<<<gWave, blk256, 0, stream>>>(bufA, as2, ad2, es, ed, N);
  gat_agg_kernel  <<<gWave, blk256, 0, stream>>>(bufA, es, ed, rowp, colx, b2, bufB, N);

  // BN + relu + mean pool
  bn_stats_kernel<<<dim3(512), blk128, 0, stream>>>(bufB, gsum, gsq, N);
  bn_pool_kernel <<<dim3(512), blk128, 0, stream>>>(bufB, batch, gsum, gsq,
                                                    bn2g, bn2b, psum, cnt, N);

  // head
  head1_kernel<<<dim3(1), blk256, 0, stream>>>(psum, cnt, doc, docW, docb, bnfg, bnfb, fbn);
  head2_kernel<<<dim3(32), blk256, 0, stream>>>(fbn, fusW, fusb, f2);
  head3_kernel<<<dim3(17), blk256, 0, stream>>>(f2, taskW, taskb, timeW, timeb,
                                                (float*)d_out);
}
